// GNODE_1176821039671
// MI455X (gfx1250) — matrix operand order repaired and checked by reading, not compiled, not run on hardware
//
#include <hip/hip_runtime.h>

#define N_NODES 200000
#define N_EDGES (N_NODES - 1)
#define RKSTEPS 4
#define TILES (N_NODES / 16)
#define DT_C 0.01f

typedef __attribute__((ext_vector_type(16))) _Float16 v16h;
typedef __attribute__((ext_vector_type(8)))  float    v8f;
typedef __attribute__((ext_vector_type(4)))  float    v4f;
typedef __attribute__((ext_vector_type(8)))  unsigned v8u;
typedef __attribute__((ext_vector_type(4)))  unsigned v4u;

__device__ __forceinline__ unsigned lane_id() {
  return (unsigned)__builtin_amdgcn_mbcnt_lo(~0u, 0u);
}

// opaque zero: defeats LICM/CSE so LDS loads stay inside loops (no reg blow-up)
__device__ __forceinline__ unsigned opq() {
  unsigned x = 0;
  asm volatile("" : "+v"(x));
  return x;
}

// swap data between lane halves (lane ^ 16) of the wave32
__device__ __forceinline__ unsigned swap16(unsigned x) {
#if __has_builtin(__builtin_amdgcn_permlanex16)
  return (unsigned)__builtin_amdgcn_permlanex16((int)x, (int)x, 0x76543210, 0xfedcba98, false, false);
#else
  return (unsigned)__builtin_amdgcn_ds_bpermute((int)((lane_id() ^ 16u) << 2), (int)x);
#endif
}

__device__ __forceinline__ unsigned pk2(float a, float b) {
  return __builtin_bit_cast(unsigned, __builtin_amdgcn_cvt_pkrtz(a, b));
}

// fast swish: x * rcp(1+exp(-x))
__device__ __forceinline__ float swish1(float x) {
  return x * __builtin_amdgcn_rcpf(1.0f + __expf(-x));
}

// Build a 32x16 f16 B-fragment (K = features, N = nodes-in-lanes) from two f32
// 16x16 C-fragments c0 (feat rows F..F+15) and c1 (rows F+16..F+31).
__device__ __forceinline__ v16h make_bfrag(v8f c0, v8f c1, bool lo) {
  v8u b;
#pragma unroll
  for (int j = 0; j < 4; ++j) {
    unsigned p0 = pk2(c0[2 * j], c0[2 * j + 1]);
    unsigned p1 = pk2(c1[2 * j], c1[2 * j + 1]);
    unsigned s0 = swap16(p0);
    unsigned s1 = swap16(p1);
    b[j]     = lo ? p0 : s1;
    b[j + 4] = lo ? s0 : p1;
  }
  return __builtin_bit_cast(v16h, b);
}

// Load a 16x32 f16 A-fragment of a weight matrix stored transposed
// row-major [out][in_pad] in LDS.  tau = 16-row tile, ch = 32-col K chunk.
__device__ __forceinline__ v16h lds_afrag(const _Float16* w, int tau, int ch,
                                          int pitch, unsigned lane) {
  int row = tau * 16 + (int)(lane & 15u);
  int k0  = ch * 32 + (int)((lane >> 4) & 1u) * 16;
  const _Float16* p = w + row * pitch + k0;
  v4u la = *(const v4u*)p;          // K parts k0..k0+7
  v4u lb = *(const v4u*)(p + 8);    // K parts k0+8..k0+15
  v8u r;
  r[0] = la[0]; r[1] = la[1]; r[2] = la[2]; r[3] = la[3];
  r[4] = lb[0]; r[5] = lb[1]; r[6] = lb[2]; r[7] = lb[3];
  return __builtin_bit_cast(v16h, r);
}

// Broadcast bias vector (feature-indexed) into a C-fragment.
__device__ __forceinline__ v8f lds_cvec(const float* b, int tau, unsigned lane) {
  const float* p = b + tau * 16 + (int)((lane >> 4) & 1u) * 8;
  v4f a = *(const v4f*)p;
  v4f c = *(const v4f*)(p + 4);
  v8f r;
  r[0] = a[0]; r[1] = a[1]; r[2] = a[2]; r[3] = a[3];
  r[4] = c[0]; r[5] = c[1]; r[6] = c[2]; r[7] = c[3];
  return r;
}

// bias1(t) = b1 + t * w1t, as a C-fragment
__device__ __forceinline__ v8f bias_t(const float* b1p, const float* wtp,
                                      int tau, float tc, unsigned lane) {
  v8f b = lds_cvec(b1p, tau, lane);
  v8f w = lds_cvec(wtp, tau, lane);
  v8f r;
#pragma unroll
  for (int v = 0; v < 8; ++v) r[v] = fmaf(tc, w[v], b[v]);
  return r;
}

__device__ __forceinline__ v8f wmma16(v16h a, v16h b, v8f c) {
  return __builtin_amdgcn_wmma_f32_16x16x32_f16(false, a, false, b, (short)0, c,
                                                false, false);
}

// Stage W (IN x OUT row-major, f32) into LDS as transposed f16 [OUTPAD][INPAD].
__device__ void stage_wT(const float* __restrict__ W, _Float16* __restrict__ dst,
                         int IN, int OUT, int OUTPAD, int INPAD, int tid, int nthr) {
  for (int idx = tid; idx < OUTPAD * INPAD; idx += nthr) {
    int o = idx / INPAD, k = idx % INPAD;
    float v = (o < OUT && k < IN) ? W[k * OUT + o] : 0.0f;
    dst[idx] = (_Float16)v;
  }
}

__device__ void stage_vec(const float* __restrict__ s, float* __restrict__ d,
                          int n, int npad, int tid, int nthr) {
  for (int i = tid; i < npad; i += nthr) d[i] = (i < n) ? s[i] : 0.0f;
}

__global__ __launch_bounds__(128, 3) void gnode_main(
    const float* __restrict__ nodes, const float* __restrict__ edges,
    const float* __restrict__ g,
    const float* __restrict__ wn1, const float* __restrict__ bn1,
    const float* __restrict__ wn2, const float* __restrict__ bn2,
    const float* __restrict__ we1, const float* __restrict__ be1,
    const float* __restrict__ we2, const float* __restrict__ be2,
    const float* __restrict__ w1,  const float* __restrict__ b1,
    const float* __restrict__ w2,  const float* __restrict__ b2,
    const float* __restrict__ wo,  const float* __restrict__ bo,
    const float* __restrict__ wd1, const float* __restrict__ bd1,
    const float* __restrict__ wd2, const float* __restrict__ bd2,
    const float* __restrict__ w3,  const float* __restrict__ b3,
    float* __restrict__ outn) {
  // ---- LDS weight cache (transposed, zero padded) ----
  __shared__ __align__(16) _Float16 sW1T[64 * 64];   // ode_w1^T   [out][in]
  __shared__ __align__(16) _Float16 sW2T[64 * 64];   // ode_w2^T
  __shared__ __align__(16) _Float16 sWOT[16 * 64];   // node_out^T
  __shared__ __align__(16) _Float16 sWN1T[16 * 32];  // enc_n_w1^T
  __shared__ __align__(16) _Float16 sWN2T[16 * 32];  // enc_n_w2^T
  __shared__ __align__(16) _Float16 sWD1T[16 * 32];  // dec_n_w1^T
  __shared__ __align__(16) _Float16 sWD2T[16 * 32];  // dec_n_w2^T
  __shared__ __align__(16) float sB1[64], sB2[64], sW1t[64];
  __shared__ __align__(16) float sBn1[16], sBn2[16], sBo[16], sBd1[16], sBd2[16];
  __shared__ __align__(16) float sWe1[16], sBe1[16], sBe2[16], sW3[16];
  __shared__ __align__(16) float sWe2[16 * 16];
  __shared__ __align__(16) float sElat[4][17 * 16];  // per-wave edge-latent buf

  const int tid = (int)threadIdx.x, nthr = (int)blockDim.x;
  stage_wT(wn1, sWN1T,  5, 16, 16, 32, tid, nthr);
  stage_wT(wn2, sWN2T, 16, 16, 16, 32, tid, nthr);
  stage_wT(w1,  sW1T,  51, 50, 64, 64, tid, nthr);
  stage_wT(w2,  sW2T,  50, 50, 64, 64, tid, nthr);
  stage_wT(wo,  sWOT,  50, 16, 16, 64, tid, nthr);
  stage_wT(wd1, sWD1T, 16, 16, 16, 32, tid, nthr);
  stage_wT(wd2, sWD2T, 16, 16, 16, 32, tid, nthr);
  stage_vec(b1, sB1, 50, 64, tid, nthr);
  stage_vec(b2, sB2, 50, 64, tid, nthr);
  for (int i = tid; i < 64; i += nthr) sW1t[i] = (i < 50) ? w1[50 * 50 + i] : 0.f;
  stage_vec(bn1, sBn1, 16, 16, tid, nthr);
  stage_vec(bn2, sBn2, 16, 16, tid, nthr);
  stage_vec(bo,  sBo,  16, 16, tid, nthr);
  stage_vec(bd1, sBd1, 16, 16, tid, nthr);
  stage_vec(bd2, sBd2, 16, 16, tid, nthr);
  stage_vec(we1, sWe1, 16, 16, tid, nthr);
  stage_vec(be1, sBe1, 16, 16, tid, nthr);
  stage_vec(be2, sBe2, 16, 16, tid, nthr);
  stage_vec(w3,  sW3,  16, 16, tid, nthr);
  for (int i = tid; i < 256; i += nthr) sWe2[i] = we2[i];
  __syncthreads();

  const unsigned lane = lane_id();
  const bool lo16 = lane < 16u;
  const int  wave = tid >> 5;
  const int  gwave = (int)blockIdx.x * (nthr >> 5) + wave;
  const int  nwaves = (int)gridDim.x * (nthr >> 5);
  float* elatBuf = sElat[wave];
  const float g0 = g[0], g1 = g[1];
  const float b3v = b3[0];
  const float hrk = 1.0f / RKSTEPS;

  // ---- resident set: layer-1 ODE weight A-fragments only (64 VGPRs) ----
  v16h A1[4][2];
#pragma unroll
  for (int tau = 0; tau < 4; ++tau) {
    A1[tau][0] = lds_afrag(sW1T, tau, 0, 64, lane);
    A1[tau][1] = lds_afrag(sW1T, tau, 1, 64, lane);
  }

  // One RK sub-step: k = f(t, X); ya += wk*k; if genNext: X <- B-frags of yb + alpha*k
  // Layer-2 weights + all biases are streamed from LDS (opaque offset stops hoisting).
  auto feval = [&](v16h& Xb0, v16h& Xb1, float tc, float wk, float alpha,
                   bool genNext, v8f (&ybl)[4], v8f (&yal)[4]) {
    unsigned off = opq();
    const _Float16* w2p = sW2T + off;
    const float* b1p = sB1 + off;
    const float* wtp = sW1t + off;
    const float* b2p = sB2 + off;
    // layer 1, H fragments built in pairs (peak 16 regs)
    v16h Hb0, Hb1;
    {
      v8f h0 = bias_t(b1p, wtp, 0, tc, lane);
      h0 = wmma16(A1[0][0], Xb0, h0);
      h0 = wmma16(A1[0][1], Xb1, h0);
      v8f h1 = bias_t(b1p, wtp, 1, tc, lane);
      h1 = wmma16(A1[1][0], Xb0, h1);
      h1 = wmma16(A1[1][1], Xb1, h1);
#pragma unroll
      for (int v = 0; v < 8; ++v) { h0[v] = fmaxf(h0[v], 0.f); h1[v] = fmaxf(h1[v], 0.f); }
      Hb0 = make_bfrag(h0, h1, lo16);
      v8f h2 = bias_t(b1p, wtp, 2, tc, lane);
      h2 = wmma16(A1[2][0], Xb0, h2);
      h2 = wmma16(A1[2][1], Xb1, h2);
      v8f h3 = bias_t(b1p, wtp, 3, tc, lane);
      h3 = wmma16(A1[3][0], Xb0, h3);
      h3 = wmma16(A1[3][1], Xb1, h3);
#pragma unroll
      for (int v = 0; v < 8; ++v) { h2[v] = fmaxf(h2[v], 0.f); h3[v] = fmaxf(h3[v], 0.f); }
      Hb1 = make_bfrag(h2, h3, lo16);
    }
    // layer 2, fused consumption of k tiles
    v8f xc0;
#pragma unroll
    for (int tau = 0; tau < 4; ++tau) {
      v8f c = lds_cvec(b2p, tau, lane);
      c = wmma16(lds_afrag(w2p, tau, 0, 64, lane), Hb0, c);
      c = wmma16(lds_afrag(w2p, tau, 1, 64, lane), Hb1, c);
#pragma unroll
      for (int v = 0; v < 8; ++v) yal[tau][v] = fmaf(wk, c[v], yal[tau][v]);
      if (genNext) {
        v8f xc;
#pragma unroll
        for (int v = 0; v < 8; ++v) xc[v] = fmaf(alpha, c[v], ybl[tau][v]);
        if ((tau & 1) == 0) {
          xc0 = xc;
        } else {
          v16h nb = make_bfrag(xc0, xc, lo16);
          if (tau == 1) Xb0 = nb; else Xb1 = nb;
        }
      }
    }
  };

#pragma unroll 1
  for (int tile = gwave; tile < TILES; tile += nwaves) {
    const int i0 = tile * 16;
    if (tile + nwaves < TILES) {
      __builtin_prefetch(nodes + (size_t)(tile + nwaves) * 16 * 5, 0, 1);
      __builtin_prefetch(edges + (size_t)(tile + nwaves) * 16, 0, 1);
    }
    const unsigned eo = opq();  // per-tile opaque offset for enc/dec LDS reads

    // ---- node features (each half-wave mirrors the 16 rows) ----
    float nf[5];
    {
      const float* np = nodes + (size_t)(i0 + (int)(lane & 15u)) * 5;
#pragma unroll
      for (int j = 0; j < 5; ++j) nf[j] = np[j];
    }
    // nodes^T as B-fragment (K = 5 features, zero padded; hi lanes K>=16 are 0)
    v8u nbu = (v8u)0u;
    nbu[0] = lo16 ? pk2(nf[0], nf[1]) : 0u;
    nbu[1] = lo16 ? pk2(nf[2], nf[3]) : 0u;
    nbu[2] = lo16 ? pk2(nf[4], 0.0f) : 0u;
    v16h nodeB = __builtin_bit_cast(v16h, nbu);

    // ---- node encoder: nlat^T = Wn2^T swish(Wn1^T nodes^T + bn1) + bn2 ----
    v8f c = lds_cvec(sBn1 + eo, 0, lane);
    c = wmma16(lds_afrag(sWN1T + eo, 0, 0, 32, lane), nodeB, c);
#pragma unroll
    for (int v = 0; v < 8; ++v) c[v] = swish1(c[v]);
    v16h hB = make_bfrag(c, (v8f)0.0f, lo16);
    v8f nlat = lds_cvec(sBn2 + eo, 0, lane);
    nlat = wmma16(lds_afrag(sWN2T + eo, 0, 0, 32, lane), hB, nlat);

    // ---- edge encoder (K=1): rows i0-1 .. i0+15, scalar per lane ----
    if (lane < 17u) {
      int e = i0 - 1 + (int)lane;
      bool valid = (e >= 0) && (e < N_EDGES);
      float ev = valid ? edges[e] : 0.0f;
      float hh[16];
#pragma unroll
      for (int i2 = 0; i2 < 16; ++i2)
        hh[i2] = swish1(fmaf(ev, sWe1[i2 + eo], sBe1[i2 + eo]));
#pragma unroll
      for (int j = 0; j < 16; ++j) {
        float a2 = sBe2[j + eo];
#pragma unroll
        for (int i2 = 0; i2 < 16; ++i2) a2 = fmaf(hh[i2], sWe2[i2 * 16 + j + eo], a2);
        elatBuf[(int)lane * 16 + j] = valid ? a2 : 0.0f;
      }
    }
    // DS ops of one wave execute in order -> buffer visible to its own reads.

    // ---- y0^T C-fragments: [nlat | sent | recv | g,0...] ----
    v8f yb[4];
    yb[0] = nlat;
    {
      int n = (int)(lane & 15u);
      int fo = (int)((lane >> 4) & 1u) * 8;
      const float* p1 = elatBuf + (n + 1) * 16 + fo;  // sent = elat[i0+n]
      const float* p2 = elatBuf + (n + 0) * 16 + fo;  // recv = elat[i0+n-1]
      v4f a = *(const v4f*)p1, bq = *(const v4f*)(p1 + 4);
      v4f d = *(const v4f*)p2, e2 = *(const v4f*)(p2 + 4);
#pragma unroll
      for (int v = 0; v < 4; ++v) {
        yb[1][v] = a[v]; yb[1][v + 4] = bq[v];
        yb[2][v] = d[v]; yb[2][v + 4] = e2[v];
      }
    }
    yb[3] = (v8f)0.0f;
    yb[3][0] = lo16 ? g0 : 0.0f;
    yb[3][1] = lo16 ? g1 : 0.0f;

    // ---- RK4 over the latent ODE ----
    float tt = 0.0f;
#pragma unroll 1
    for (int step = 0; step < RKSTEPS; ++step) {
      v8f ya[4];
#pragma unroll
      for (int tau = 0; tau < 4; ++tau) ya[tau] = yb[tau];
      v16h Xb0 = make_bfrag(yb[0], yb[1], lo16);
      v16h Xb1 = make_bfrag(yb[2], yb[3], lo16);
      feval(Xb0, Xb1, tt,              hrk / 6.0f, 0.5f * hrk, true,  yb, ya);  // k1
      feval(Xb0, Xb1, tt + 0.5f * hrk, hrk / 3.0f, 0.5f * hrk, true,  yb, ya);  // k2
      feval(Xb0, Xb1, tt + 0.5f * hrk, hrk / 3.0f, hrk,        true,  yb, ya);  // k3
      feval(Xb0, Xb1, tt + hrk,        hrk / 6.0f, 0.0f,       false, yb, ya);  // k4
#pragma unroll
      for (int tau = 0; tau < 4; ++tau) yb[tau] = ya[tau];
      tt += hrk;
    }

    // ---- output projection + decoder MLP ----
    v16h Yb0 = make_bfrag(yb[0], yb[1], lo16);
    v16h Yb1 = make_bfrag(yb[2], yb[3], lo16);
    c = lds_cvec(sBo + eo, 0, lane);
    c = wmma16(lds_afrag(sWOT + eo, 0, 0, 64, lane), Yb0, c);
    c = wmma16(lds_afrag(sWOT + eo, 0, 1, 64, lane), Yb1, c);   // nlat2^T (linear)
    v16h dB = make_bfrag(c, (v8f)0.0f, lo16);
    c = lds_cvec(sBd1 + eo, 0, lane);
    c = wmma16(lds_afrag(sWD1T + eo, 0, 0, 32, lane), dB, c);
#pragma unroll
    for (int v = 0; v < 8; ++v) c[v] = swish1(c[v]);
    dB = make_bfrag(c, (v8f)0.0f, lo16);
    c = lds_cvec(sBd2 + eo, 0, lane);
    c = wmma16(lds_afrag(sWD2T + eo, 0, 0, 32, lane), dB, c);
#pragma unroll
    for (int v = 0; v < 8; ++v) c[v] = swish1(c[v]);

    // pred_acc[node] = dot(dh2[node, :16], w3) + b3  (halves hold feats 0-7/8-15)
    float part = 0.0f;
    {
      int fo = (int)((lane >> 4) & 1u) * 8;
#pragma unroll
      for (int v = 0; v < 8; ++v) part = fmaf(c[v], sW3[fo + v + eo], part);
    }
    float other = __builtin_bit_cast(float, swap16(__builtin_bit_cast(unsigned, part)));
    float acc = part + other + b3v;

    if (lo16) {
      int i = i0 + (int)lane;
      float nv = fmaf(acc, DT_C, nf[4]);
      float np = fmaf(nv, DT_C, nf[0]);
      float* o = outn + (size_t)i * 6;
      o[0] = np; o[1] = nf[2]; o[2] = nf[3]; o[3] = nf[4]; o[4] = nv; o[5] = acc;
    }
  }
}

__global__ void gnode_fin(const float* __restrict__ outn, float* __restrict__ oute,
                          float* __restrict__ outg, const float* __restrict__ g) {
  int i = (int)blockIdx.x * (int)blockDim.x + (int)threadIdx.x;
  if (i < N_EDGES)
    oute[i] = outn[(size_t)(i + 1) * 6] - outn[(size_t)i * 6];
  if (i == 0) { outg[0] = g[0] + 1.0f; outg[1] = g[1]; }
}

extern "C" void kernel_launch(void* const* d_in, const int* in_sizes, int n_in,
                              void* d_out, int out_size, void* d_ws, size_t ws_size,
                              hipStream_t stream) {
  (void)in_sizes; (void)n_in; (void)out_size; (void)d_ws; (void)ws_size;
  const float* nodes = (const float*)d_in[0];
  const float* edges = (const float*)d_in[1];
  const float* g     = (const float*)d_in[2];
  const float* wn1   = (const float*)d_in[3];
  const float* bn1   = (const float*)d_in[4];
  const float* wn2   = (const float*)d_in[5];
  const float* bn2   = (const float*)d_in[6];
  const float* we1   = (const float*)d_in[7];
  const float* be1   = (const float*)d_in[8];
  const float* we2   = (const float*)d_in[9];
  const float* be2   = (const float*)d_in[10];
  const float* w1    = (const float*)d_in[11];
  const float* b1    = (const float*)d_in[12];
  const float* w2    = (const float*)d_in[13];
  const float* b2    = (const float*)d_in[14];
  const float* wo    = (const float*)d_in[15];
  const float* bo    = (const float*)d_in[16];
  const float* wd1   = (const float*)d_in[17];
  const float* bd1   = (const float*)d_in[18];
  const float* wd2   = (const float*)d_in[19];
  const float* bd2   = (const float*)d_in[20];
  const float* w3    = (const float*)d_in[21];
  const float* b3    = (const float*)d_in[22];

  float* out       = (float*)d_out;
  float* out_nodes = out;
  float* out_edges = out + (size_t)N_NODES * 6;
  float* out_g     = out_edges + N_EDGES;

  gnode_main<<<dim3(1024), dim3(128), 0, stream>>>(
      nodes, edges, g, wn1, bn1, wn2, bn2, we1, be1, we2, be2,
      w1, b1, w2, b2, wo, bo, wd1, bd1, wd2, bd2, w3, b3, out_nodes);

  int thr = 256;
  int nb = (N_EDGES + thr - 1) / thr;
  gnode_fin<<<dim3(nb), dim3(thr), 0, stream>>>(out_nodes, out_edges, out_g, g);
}
